// SelfAttention_21165598835575
// MI455X (gfx1250) — matrix operand order repaired and checked by reading, not compiled, not run on hardware
//
#include <hip/hip_runtime.h>

// ---------------------------------------------------------------------------
// Types / helpers
// ---------------------------------------------------------------------------
typedef __attribute__((ext_vector_type(16))) __bf16 bf16x16;
typedef __attribute__((ext_vector_type(8)))  float  f32x8;

union FragU { bf16x16 v; uint4 q[2]; };
union U4S8  { uint4 q; unsigned short s[8]; };

#if defined(__has_builtin)
#if __has_builtin(__builtin_amdgcn_global_load_async_to_lds_b128) && \
    __has_builtin(__builtin_amdgcn_s_wait_asynccnt)
#define USE_ASYNC_LDS 1
#endif
#endif
#ifndef USE_ASYNC_LDS
#define USE_ASYNC_LDS 0
#endif

// 16-byte int vector pointer types matching the async-LDS builtin signature
typedef int v4i_vs __attribute__((vector_size(16)));
typedef __attribute__((address_space(1))) v4i_vs* gp128_t;   // global
typedef __attribute__((address_space(3))) v4i_vs* lp128_t;   // LDS

// copy 16 bytes global -> LDS (per lane).  Async path uses the CDNA5
// GLOBAL_LOAD_ASYNC_TO_LDS_B128 DMA (ASYNCcnt), bypassing VGPRs.
__device__ __forceinline__ void cp_b128(void* lds, const void* g) {
#if USE_ASYNC_LDS
    __builtin_amdgcn_global_load_async_to_lds_b128(
        (gp128_t)(void*)g, (lp128_t)lds, 0, 0);
#else
    *(uint4*)lds = *(const uint4*)g;
#endif
}
__device__ __forceinline__ void cp_wait() {
#if USE_ASYNC_LDS
    __builtin_amdgcn_s_wait_asynccnt(0);
#endif
}

__device__ __forceinline__ unsigned short f32_to_bf16_bits(float f) {
    unsigned u = __builtin_bit_cast(unsigned, f);
    u += 0x7fffu + ((u >> 16) & 1u);          // round to nearest even
    return (unsigned short)(u >> 16);
}

// A fragment: 16x32 bf16 tile (MxK), row-major source with leading dim ld.
__device__ __forceinline__ bf16x16 frag_a(const unsigned short* base, int row0, int ld, int k0) {
    int lane = threadIdx.x & 31;
    int m    = lane & 15;
    int kb   = (lane >> 4) << 3;              // 0 or 8
    const unsigned short* p = base + (size_t)(row0 + m) * ld + k0 + kb;
    FragU f;
    f.q[0] = *(const uint4*)(p);              // K = kb+0 .. kb+7
    f.q[1] = *(const uint4*)(p + 16);         // K = kb+16 .. kb+23
    return f.v;
}

// B fragment: 32x16 bf16 tile (KxN) from B^T stored row-major (N rows, K cols).
__device__ __forceinline__ bf16x16 frag_b(const unsigned short* base, int n0, int ld, int k0) {
    int lane = threadIdx.x & 31;
    int n    = lane & 15;
    int kb   = (lane >> 4) << 3;              // 0 or 8
    const unsigned short* p = base + (size_t)(n0 + n) * ld + k0 + kb;
    FragU f;
    f.q[0] = *(const uint4*)(p);
    f.q[1] = *(const uint4*)(p + 16);
    return f.v;
}

__device__ __forceinline__ f32x8 wmma_bf16(bf16x16 a, bf16x16 b, f32x8 c) {
    return __builtin_amdgcn_wmma_f32_16x16x32_bf16(false, a, false, b, (short)0, c, false, false);
}

// ---------------------------------------------------------------------------
// Problem constants
// ---------------------------------------------------------------------------
#define BATCH 2
#define SEQ   2048
#define HDIM  512
#define NHEAD 8
#define MDIM  (BATCH * SEQ)        // 4096
#define NQKV  (NHEAD * HDIM)       // 4096
#define SOFT_SCALE 0.04419417382415922f   // 512^-0.5

// ---------------------------------------------------------------------------
// fp32 -> bf16 conversion
// ---------------------------------------------------------------------------
__global__ void cvt_bf16_kernel(const float* __restrict__ in, unsigned short* __restrict__ out, int n) {
    int i = blockIdx.x * blockDim.x + threadIdx.x;
    if (i < n) out[i] = f32_to_bf16_bits(in[i]);
}

// ---------------------------------------------------------------------------
// QKV projection GEMM (NT): C[m,n] = sum_d x[m,d] * W[n,d]
// M=4096, N=4096, K=512.  Block tile 128x128, 8 waves (4 along M x 2 along N),
// each wave computes 32x64 via 2x4 WMMA tiles (8 WMMA per k-step).
// Output remapped to [b,h,s,e] bf16.  blockIdx.z picks Wq / Wk / Wv.
// ---------------------------------------------------------------------------
__global__ __launch_bounds__(256) void gemm_qkv_kernel(
    const unsigned short* __restrict__ xb,
    const unsigned short* __restrict__ wqb,
    const unsigned short* __restrict__ wkb,
    const unsigned short* __restrict__ wvb,
    unsigned short* __restrict__ qb,
    unsigned short* __restrict__ kbuf,
    unsigned short* __restrict__ vb) {
    __shared__ unsigned short As[128 * 32];   // 8 KB
    __shared__ unsigned short Bs[128 * 32];   // 8 KB

    const unsigned short* W = (blockIdx.z == 0) ? wqb : (blockIdx.z == 1) ? wkb : wvb;
    unsigned short*       O = (blockIdx.z == 0) ? qb  : (blockIdx.z == 1) ? kbuf : vb;

    const int tid = threadIdx.x, wave = tid >> 5, lane = tid & 31;
    const int wm = wave >> 1, wn = wave & 1;
    const int m0 = blockIdx.x * 128, n0 = blockIdx.y * 128;

    f32x8 acc[2][4] = {};

    for (int k0 = 0; k0 < HDIM; k0 += 32) {
        __syncthreads();
        // A tile 128x32 and B tile 128x32: 512 uint4 each, 2 per thread
        #pragma unroll
        for (int ii = 0; ii < 2; ++ii) {
            int i = tid + ii * 256;
            int r = i >> 2, c = (i & 3) << 3;
            cp_b128(&As[r * 32 + c], &xb[(size_t)(m0 + r) * HDIM + k0 + c]);
            cp_b128(&Bs[r * 32 + c], &W[(size_t)(n0 + r) * HDIM + k0 + c]);
        }
        if (k0 + 32 < HDIM) {           // prefetch next tiles into L2/L0
            int r = tid & 127, cc = (tid >> 7) << 4;
            __builtin_prefetch(&xb[(size_t)(m0 + r) * HDIM + k0 + 32 + cc], 0, 3);
            __builtin_prefetch(&W[(size_t)(n0 + r) * HDIM + k0 + 32 + cc], 0, 3);
        }
        cp_wait();
        __syncthreads();

        bf16x16 a0 = frag_a(As, wm * 32 + 0,  32, 0);
        bf16x16 a1 = frag_a(As, wm * 32 + 16, 32, 0);
        bf16x16 b0 = frag_b(Bs, wn * 64 + 0,  32, 0);
        bf16x16 b1 = frag_b(Bs, wn * 64 + 16, 32, 0);
        bf16x16 b2 = frag_b(Bs, wn * 64 + 32, 32, 0);
        bf16x16 b3 = frag_b(Bs, wn * 64 + 48, 32, 0);
        acc[0][0] = wmma_bf16(a0, b0, acc[0][0]);
        acc[0][1] = wmma_bf16(a0, b1, acc[0][1]);
        acc[0][2] = wmma_bf16(a0, b2, acc[0][2]);
        acc[0][3] = wmma_bf16(a0, b3, acc[0][3]);
        acc[1][0] = wmma_bf16(a1, b0, acc[1][0]);
        acc[1][1] = wmma_bf16(a1, b1, acc[1][1]);
        acc[1][2] = wmma_bf16(a1, b2, acc[1][2]);
        acc[1][3] = wmma_bf16(a1, b3, acc[1][3]);
    }

    // epilogue: remap (m,n) -> [b, h, s, e] bf16
    const int nn = lane & 15, hi = lane >> 4;
    for (int mi = 0; mi < 2; ++mi)
        for (int ni = 0; ni < 4; ++ni)
            for (int j = 0; j < 8; ++j) {
                int m = m0 + wm * 32 + mi * 16 + hi * 8 + j;
                int n = n0 + wn * 64 + ni * 16 + nn;
                int h = n >> 9, e = n & 511;
                int bb = m >> 11, s = m & 2047;
                O[(((size_t)bb * NHEAD + h) * SEQ + s) * HDIM + e] = f32_to_bf16_bits(acc[mi][ni][j]);
            }
}

// ---------------------------------------------------------------------------
// Flash attention.  One block per (b, h, 32-row query block).  8 waves:
// wave = strip*4 + cchunk ; strip in {0,1} (16 query rows), cchunk in {0..3}
// (128-wide slice of head dim).  Online softmax with causal mask.
// ---------------------------------------------------------------------------
__global__ __launch_bounds__(256) void attention_kernel(
    const unsigned short* __restrict__ qbuf,
    const unsigned short* __restrict__ kbuf,
    const unsigned short* __restrict__ vbuf,
    unsigned short* __restrict__ hout) {
    extern __shared__ char smem_raw[];
    unsigned short* Qs    = (unsigned short*)smem_raw;          // 32 x 512 bf16 (32 KB)
    unsigned short* Ks    = Qs + 32 * 512;                      // 32 x 512 bf16 (32 KB)
    unsigned short* Vts   = Ks + 32 * 512;                      // 512 x 32 bf16, transposed (32 KB)
    float*          Spart = (float*)(Vts + 32 * 512);           // 8 waves x 16 x 32 f32 (16 KB)
    unsigned short* Ps    = (unsigned short*)(Spart + 8 * 512); // 32 x 32 bf16 (2 KB)
    float*          rowM  = (float*)(Ps + 32 * 32);
    float*          rowL  = rowM + 32;
    float*          rowScale = rowL + 32;

    const int bx   = blockIdx.x;
    const int sblk = bx & 63;
    const int h    = (bx >> 6) & 7;
    const int bb   = bx >> 9;
    const int s0   = sblk * 32;
    const size_t bh = (size_t)(bb * NHEAD + h);

    const unsigned short* qg = qbuf + bh * SEQ * HDIM;
    const unsigned short* kg = kbuf + bh * SEQ * HDIM;
    const unsigned short* vg = vbuf + bh * SEQ * HDIM;

    const int tid = threadIdx.x, wave = tid >> 5, lane = tid & 31;
    const int strip = wave >> 2, cchunk = wave & 3;
    const int e0 = cchunk * 128;

    // load Q block (32 x 512) into LDS (async DMA)
    #pragma unroll
    for (int ii = 0; ii < 8; ++ii) {
        int i = tid + ii * 256;
        int r = i >> 6, c = (i & 63) << 3;
        cp_b128(&Qs[r * 512 + c], &qg[(size_t)(s0 + r) * HDIM + c]);
    }
    if (tid < 32) { rowM[tid] = -1e30f; rowL[tid] = 0.0f; }

    f32x8 acc[8] = {};   // 16 rows x 128 cols of O for this wave

    const int nKTiles = sblk + 1;   // causal: only key tiles with t0 <= s0
    for (int it = 0; it < nKTiles; ++it) {
        const int t0 = it * 32;
        __syncthreads();   // previous P@V done reading Ks/Vts

        // K tile row-major via async DMA
        #pragma unroll
        for (int ii = 0; ii < 8; ++ii) {
            int i = tid + ii * 256;
            int r = i >> 6, c = (i & 63) << 3;
            cp_b128(&Ks[r * 512 + c], &kg[(size_t)(t0 + r) * HDIM + c]);
        }
        // V tile transposed [e][t]: batch global loads, then scatter to LDS
        {
            uint4 vtmp[8];
            #pragma unroll
            for (int ii = 0; ii < 8; ++ii) {
                int i = tid + ii * 256;
                int r = i >> 6, c = (i & 63) << 3;
                vtmp[ii] = *(const uint4*)(&vg[(size_t)(t0 + r) * HDIM + c]);
            }
            #pragma unroll
            for (int ii = 0; ii < 8; ++ii) {
                int i = tid + ii * 256;
                int r = i >> 6, c = (i & 63) << 3;
                U4S8 u; u.q = vtmp[ii];
                #pragma unroll
                for (int j = 0; j < 8; ++j) Vts[(c + j) * 32 + r] = u.s[j];
            }
        }
        if (it + 1 < nKTiles) {   // prefetch next K/V tiles
            int r = tid >> 3, c = (tid & 7) << 6;
            __builtin_prefetch(&kg[(size_t)(t0 + 32 + r) * HDIM + c], 0, 3);
            __builtin_prefetch(&vg[(size_t)(t0 + 32 + r) * HDIM + c], 0, 3);
        }
        cp_wait();
        __syncthreads();

        // partial S = Q(strip rows, e-chunk) . K^T  -> 16x32 f32
        {
            f32x8 sacc[2] = {};
            #pragma unroll
            for (int kk = 0; kk < 128; kk += 32) {
                bf16x16 a  = frag_a(Qs, strip * 16, 512, e0 + kk);
                bf16x16 b0 = frag_b(Ks, 0,  512, e0 + kk);
                bf16x16 b1 = frag_b(Ks, 16, 512, e0 + kk);
                sacc[0] = wmma_bf16(a, b0, sacc[0]);
                sacc[1] = wmma_bf16(a, b1, sacc[1]);
            }
            float* sp = Spart + wave * 512;
            int n = lane & 15, hi = lane >> 4;
            #pragma unroll
            for (int nt = 0; nt < 2; ++nt)
                #pragma unroll
                for (int j = 0; j < 8; ++j)
                    sp[(hi * 8 + j) * 32 + nt * 16 + n] = sacc[nt][j];
        }
        __syncthreads();

        // online softmax: thread r handles query row r (0..31)
        if (tid < 32) {
            const int r = tid, g = s0 + r;
            const int st = r >> 4, mr = r & 15;
            float mold = rowM[r], mmax = mold;
            float vals[32];
            #pragma unroll
            for (int j = 0; j < 32; ++j) {
                float v = 0.0f;
                #pragma unroll
                for (int c = 0; c < 4; ++c)
                    v += Spart[(size_t)(st * 4 + c) * 512 + mr * 32 + j];
                v *= SOFT_SCALE;
                if (t0 + j > g) v = -1e30f;
                vals[j] = v;
                mmax = fmaxf(mmax, v);
            }
            float fac = __expf(mold - mmax);
            float l = rowL[r] * fac;
            #pragma unroll
            for (int j = 0; j < 32; ++j) {
                float p = __expf(vals[j] - mmax);
                l += p;
                Ps[r * 32 + j] = f32_to_bf16_bits(p);
            }
            rowScale[r] = fac; rowM[r] = mmax; rowL[r] = l;
        }
        __syncthreads();

        // rescale accumulator and O += P . V(chunk)
        {
            int hi = lane >> 4;
            float fac[8];
            #pragma unroll
            for (int j = 0; j < 8; ++j) fac[j] = rowScale[strip * 16 + hi * 8 + j];
            #pragma unroll
            for (int nt = 0; nt < 8; ++nt)
                #pragma unroll
                for (int j = 0; j < 8; ++j) acc[nt][j] *= fac[j];

            bf16x16 pa = frag_a(Ps, strip * 16, 32, 0);
            #pragma unroll
            for (int nt = 0; nt < 8; ++nt) {
                bf16x16 bv = frag_b(Vts, e0 + nt * 16, 32, 0);
                acc[nt] = wmma_bf16(pa, bv, acc[nt]);
            }
        }
    }
    __syncthreads();

    // epilogue: O / l, store bf16 heads_out[b, s, h*512 + e]
    {
        int n = lane & 15, hi = lane >> 4;
        float linv[8];
        #pragma unroll
        for (int j = 0; j < 8; ++j) linv[j] = 1.0f / rowL[strip * 16 + hi * 8 + j];
        #pragma unroll
        for (int nt = 0; nt < 8; ++nt)
            #pragma unroll
            for (int j = 0; j < 8; ++j) {
                int srow = s0 + strip * 16 + hi * 8 + j;
                size_t idx = ((size_t)bb * SEQ + srow) * NQKV + h * HDIM + e0 + nt * 16 + n;
                hout[idx] = f32_to_bf16_bits(acc[nt][j] * linv[j]);
            }
    }
}

// ---------------------------------------------------------------------------
// Output projection GEMM (NT): out[m,n] = sum_d ho[m,d] * Wu[n,d]
// M=4096, N=512, K=4096.  Same 128x128 tiling; f32 output.
// ---------------------------------------------------------------------------
__global__ __launch_bounds__(256) void gemm_out_kernel(
    const unsigned short* __restrict__ ho,
    const unsigned short* __restrict__ wub,
    float* __restrict__ out) {
    __shared__ unsigned short As[128 * 32];
    __shared__ unsigned short Bs[128 * 32];

    const int tid = threadIdx.x, wave = tid >> 5, lane = tid & 31;
    const int wm = wave >> 1, wn = wave & 1;
    const int m0 = blockIdx.x * 128, n0 = blockIdx.y * 128;
    const int Kd = NQKV;   // 4096

    f32x8 acc[2][4] = {};

    for (int k0 = 0; k0 < Kd; k0 += 32) {
        __syncthreads();
        #pragma unroll
        for (int ii = 0; ii < 2; ++ii) {
            int i = tid + ii * 256;
            int r = i >> 2, c = (i & 3) << 3;
            cp_b128(&As[r * 32 + c], &ho[(size_t)(m0 + r) * Kd + k0 + c]);
            cp_b128(&Bs[r * 32 + c], &wub[(size_t)(n0 + r) * Kd + k0 + c]);
        }
        if (k0 + 32 < Kd) {
            int r = tid & 127, cc = (tid >> 7) << 4;
            __builtin_prefetch(&ho[(size_t)(m0 + r) * Kd + k0 + 32 + cc], 0, 3);
            __builtin_prefetch(&wub[(size_t)(n0 + r) * Kd + k0 + 32 + cc], 0, 3);
        }
        cp_wait();
        __syncthreads();

        bf16x16 a0 = frag_a(As, wm * 32 + 0,  32, 0);
        bf16x16 a1 = frag_a(As, wm * 32 + 16, 32, 0);
        bf16x16 b0 = frag_b(Bs, wn * 64 + 0,  32, 0);
        bf16x16 b1 = frag_b(Bs, wn * 64 + 16, 32, 0);
        bf16x16 b2 = frag_b(Bs, wn * 64 + 32, 32, 0);
        bf16x16 b3 = frag_b(Bs, wn * 64 + 48, 32, 0);
        acc[0][0] = wmma_bf16(a0, b0, acc[0][0]);
        acc[0][1] = wmma_bf16(a0, b1, acc[0][1]);
        acc[0][2] = wmma_bf16(a0, b2, acc[0][2]);
        acc[0][3] = wmma_bf16(a0, b3, acc[0][3]);
        acc[1][0] = wmma_bf16(a1, b0, acc[1][0]);
        acc[1][1] = wmma_bf16(a1, b1, acc[1][1]);
        acc[1][2] = wmma_bf16(a1, b2, acc[1][2]);
        acc[1][3] = wmma_bf16(a1, b3, acc[1][3]);
    }

    const int nn = lane & 15, hi = lane >> 4;
    for (int mi = 0; mi < 2; ++mi)
        for (int ni = 0; ni < 4; ++ni)
            for (int j = 0; j < 8; ++j) {
                int m = m0 + wm * 32 + mi * 16 + hi * 8 + j;
                int n = n0 + wn * 64 + ni * 16 + nn;
                out[(size_t)m * HDIM + n] = acc[mi][ni][j];
            }
}

// ---------------------------------------------------------------------------
// Launcher
// ---------------------------------------------------------------------------
extern "C" void kernel_launch(void* const* d_in, const int* in_sizes, int n_in,
                              void* d_out, int out_size, void* d_ws, size_t ws_size,
                              hipStream_t stream) {
    (void)in_sizes; (void)n_in; (void)out_size; (void)ws_size;
    const float* x  = (const float*)d_in[0];
    const float* Wq = (const float*)d_in[1];
    const float* Wk = (const float*)d_in[2];
    const float* Wv = (const float*)d_in[3];
    const float* Wu = (const float*)d_in[4];
    float* out = (float*)d_out;

    const size_t NX  = (size_t)BATCH * SEQ * HDIM;          // 2,097,152
    const size_t NW  = (size_t)NQKV * HDIM;                 // 2,097,152
    const size_t NBH = (size_t)BATCH * NHEAD * SEQ * HDIM;  // 16,777,216

    unsigned short* ws  = (unsigned short*)d_ws;
    unsigned short* xb  = ws;
    unsigned short* wqb = xb  + NX;
    unsigned short* wkb = wqb + NW;
    unsigned short* wvb = wkb + NW;
    unsigned short* wub = wvb + NW;
    unsigned short* qb  = wub + NW;
    unsigned short* kb  = qb  + NBH;
    unsigned short* vb  = kb  + NBH;
    unsigned short* hob = vb  + NBH;

    // 1) convert everything to bf16
    const int CVT_BLK = 256;
    cvt_bf16_kernel<<<(int)((NX + CVT_BLK - 1) / CVT_BLK), CVT_BLK, 0, stream>>>(x,  xb,  (int)NX);
    cvt_bf16_kernel<<<(int)((NW + CVT_BLK - 1) / CVT_BLK), CVT_BLK, 0, stream>>>(Wq, wqb, (int)NW);
    cvt_bf16_kernel<<<(int)((NW + CVT_BLK - 1) / CVT_BLK), CVT_BLK, 0, stream>>>(Wk, wkb, (int)NW);
    cvt_bf16_kernel<<<(int)((NW + CVT_BLK - 1) / CVT_BLK), CVT_BLK, 0, stream>>>(Wv, wvb, (int)NW);
    cvt_bf16_kernel<<<(int)((NW + CVT_BLK - 1) / CVT_BLK), CVT_BLK, 0, stream>>>(Wu, wub, (int)NW);

    // 2) QKV projections: [4096 x 4096] = x[4096 x 512] . W^T
    gemm_qkv_kernel<<<dim3(MDIM / 128, NQKV / 128, 3), 256, 0, stream>>>(
        xb, wqb, wkb, wvb, qb, kb, vb);

    // 3) flash attention: 2*8*64 = 1024 blocks, ~114 KB dynamic LDS
    const size_t attn_lds = (size_t)(3 * 32 * 512) * 2   // Qs, Ks, Vts
                          + (size_t)(8 * 512) * 4        // Spart
                          + (size_t)(32 * 32) * 2        // Ps
                          + (size_t)(3 * 32) * 4;        // rowM/rowL/rowScale
    attention_kernel<<<BATCH * NHEAD * (SEQ / 32), 256, attn_lds, stream>>>(qb, kb, vb, hob);

    // 4) output projection: [4096 x 512] = heads_out[4096 x 4096] . Wu^T
    gemm_out_kernel<<<dim3(MDIM / 128, HDIM / 128), 256, 0, stream>>>(hob, wub, out);
}
